// KSAFusionModule_53017076302570
// MI455X (gfx1250) — matrix operand order repaired and checked by reading, not compiled, not run on hardware
//
#include <hip/hip_runtime.h>
#include <math.h>

// ---------------------------------------------------------------------------
// Model constants (from the reference)
// ---------------------------------------------------------------------------
#define BB     8
#define CC     768
#define HH     32
#define WW2    32
#define TT     1024          // H*W
#define LNUM   2
#define FFD    3072
#define NHEAD  8
#define HD     96            // C / NHEAD
#define NPTS   2048          // PH*PW*K
#define TOPK5  5
#define ARND   5
#define NMOD   2

typedef __attribute__((ext_vector_type(16))) _Float16 v16h;
typedef __attribute__((ext_vector_type(8)))  _Float16 v8h;
typedef __attribute__((ext_vector_type(8)))  float    v8f;

// ---------------------------------------------------------------------------
// CDNA5 async global->LDS staging (guarded: falls back to manual copies)
// Builtin signature (from hipcc diagnostic): first param is
// 'int __vector(4) __device__ *'  ->  (global v4i*, local v4i*, imm, imm)
// ---------------------------------------------------------------------------
#if __has_builtin(__builtin_amdgcn_global_load_async_to_lds_b128) && \
    __has_builtin(__builtin_amdgcn_s_wait_asynccnt)
#define HAS_ASYNC 1
#else
#define HAS_ASYNC 0
#endif

#if HAS_ASYNC
typedef int v4i_ __attribute__((ext_vector_type(4)));
__device__ __forceinline__ void async_g2l_b128(const void* g, void* l) {
  __builtin_amdgcn_global_load_async_to_lds_b128(
      (__attribute__((address_space(1))) v4i_*)(unsigned long long)g,
      (__attribute__((address_space(3))) v4i_*)(unsigned long long)l,
      0, 0);
}
__device__ __forceinline__ void async_wait0() {
  __builtin_amdgcn_s_wait_asynccnt(0);
}
#else
__device__ __forceinline__ void async_wait0() {}
#endif

// ---------------------------------------------------------------------------
// Block reduction helpers (blockDim.x == 256 assumed)
// ---------------------------------------------------------------------------
__device__ __forceinline__ float blk_sum(float v, float* red) {
  int tid = threadIdx.x;
  red[tid] = v; __syncthreads();
  #pragma unroll
  for (int s = 128; s > 0; s >>= 1) {
    if (tid < s) red[tid] += red[tid + s];
    __syncthreads();
  }
  float r = red[0]; __syncthreads();
  return r;
}

__device__ __forceinline__ float blk_max(float v, float* red) {
  int tid = threadIdx.x;
  red[tid] = v; __syncthreads();
  #pragma unroll
  for (int s = 128; s > 0; s >>= 1) {
    if (tid < s) red[tid] = fmaxf(red[tid], red[tid + s]);
    __syncthreads();
  }
  float r = red[0]; __syncthreads();
  return r;
}

// ---------------------------------------------------------------------------
// Bilinear sampling (torch grid_sample: bilinear, zero pad, align_corners=F)
// ---------------------------------------------------------------------------
struct Bilin {
  int   off[4];
  float w[4];
};

__device__ __forceinline__ Bilin bilin_setup(float gx, float gy, int Hh, int Ww) {
  float ix = ((gx + 1.f) * (float)Ww - 1.f) * 0.5f;
  float iy = ((gy + 1.f) * (float)Hh - 1.f) * 0.5f;
  float fx0 = floorf(ix), fy0 = floorf(iy);
  float wx = ix - fx0, wy = iy - fy0;
  int x0 = (int)fx0, y0 = (int)fy0;
  Bilin r;
  #pragma unroll
  for (int q = 0; q < 4; ++q) {
    int dx = q & 1, dy = q >> 1;
    int x = x0 + dx, y = y0 + dy;
    bool valid = (x >= 0) && (x < Ww) && (y >= 0) && (y < Hh);
    int xc = min(max(x, 0), Ww - 1);
    int yc = min(max(y, 0), Hh - 1);
    r.off[q] = yc * Ww + xc;
    float wq = (dx ? wx : 1.f - wx) * (dy ? wy : 1.f - wy);
    r.w[q] = valid ? wq : 0.f;
  }
  return r;
}

// ---------------------------------------------------------------------------
// f32 -> f16 conversion (n must be a multiple of 4)
// ---------------------------------------------------------------------------
__global__ __launch_bounds__(256)
void cvt_h_kernel(const float* __restrict__ in, _Float16* __restrict__ out,
                  long long n)
{
  for (long long i = ((long long)blockIdx.x * 256 + threadIdx.x) * 4; i < n;
       i += (long long)gridDim.x * 256 * 4) {
    #pragma unroll
    for (int j = 0; j < 4; ++j) out[i + j] = (_Float16)in[i + j];
  }
}

// ---------------------------------------------------------------------------
// Generic batched WMMA GEMM on f16 operands, f32 accumulate.
//   C = act(A[M,K] @ B[K,N] + bias)
//   BT=1  : B stored [N,K] row-major -> computes A @ B^T
//   ACT=1 : ReLU
//   OUTH=1: store C as f16, else f32
// Block tile 32(M) x 64(N), K-step 32; 8 waves, one 16x16 WMMA tile each.
// LDS: A tile [M=32][K=32] packed; B tile stored [N=64][K=32] (stride 40)
// so every lane's WMMA fragment is two contiguous ds_load_b128.
// ---------------------------------------------------------------------------
template<int BT, int ACT, int OUTH>
__global__ __launch_bounds__(256)
void gemm_wmma(const _Float16* __restrict__ A, int lda, long long sAb, long long sAh,
               const _Float16* __restrict__ Bm, int ldb, long long sBb, long long sBh,
               void* __restrict__ Cv, int ldc, long long sCb, long long sCh,
               const float* __restrict__ bias, int M, int N, int K, int HDIV)
{
  __shared__ __align__(16) _Float16 As[32][32];
  __shared__ __align__(16) _Float16 Bs[64][40];

  int z = blockIdx.z;
  int zb = z / HDIV, zh = z % HDIV;
  A  += zb * sAb + zh * sAh;
  Bm += zb * sBb + zh * sBh;
  long long cOff = (long long)zb * sCb + (long long)zh * sCh;

  int mBase = blockIdx.y * 32;
  int nBase = blockIdx.x * 64;
  int tid  = threadIdx.x;
  int lane = tid & 31;
  int wave = tid >> 5;
  int mw = (wave & 1) * 16;
  int nw = (wave >> 1) * 16;
  int lm = lane & 15;
  int lh = lane >> 4;

  bool fullM = (mBase + 32 <= M);
  bool fullN = (nBase + 64 <= N);
  v8h vz = {};

  // staging coordinates
  int ar = tid >> 2, ac8 = (tid & 3) * 8;           // A: threads < 128
  int btn = tid >> 2, btk8 = (tid & 3) * 8;         // B (BT=1)
  int bk  = tid >> 3, bn8  = (tid & 7) * 8;         // B (BT=0)

  v8f acc = {};

  for (int k0 = 0; k0 < K; k0 += 32) {
    // ---- stage A tile: 32 rows x 32 halves = 128 x 16B chunks -------------
    if (tid < 128) {
      int gm = mBase + ar;
      const _Float16* gp = A + (long long)min(gm, M - 1) * lda + (k0 + ac8);
#if HAS_ASYNC
      if (fullM) {
        async_g2l_b128(gp, &As[ar][ac8]);
      } else
#endif
      {
        v8h v = *(const v8h*)gp;
        if (gm >= M) v = vz;
        *(v8h*)&As[ar][ac8] = v;
      }
    }
    // ---- stage B tile into [N][K] layout ----------------------------------
    if (BT) {
      int gn = nBase + btn;
      const _Float16* gp = Bm + (long long)min(gn, N - 1) * ldb + (k0 + btk8);
#if HAS_ASYNC
      if (fullN) {
        async_g2l_b128(gp, &Bs[btn][btk8]);
      } else
#endif
      {
        v8h v = *(const v8h*)gp;
        if (gn >= N) v = vz;
        *(v8h*)&Bs[btn][btk8] = v;
      }
    } else {
      int gn = nBase + bn8;
      v8h v = *(const v8h*)(Bm + (long long)(k0 + bk) * ldb + min(gn, N - 8));
      if (gn + 8 > N) v = vz;
      #pragma unroll
      for (int j = 0; j < 8; ++j) Bs[bn8 + j][bk] = v[j];
    }
    async_wait0();
    __syncthreads();

    // ---- fragments: two contiguous b128 LDS loads each --------------------
    // A lane (m=mw+lm): elems e    -> K = 16*lh+e
    // B lane (n=nw+lm): elems e    -> K = 16*lh+e
    v8h a0 = *(const v8h*)&As[mw + lm][16 * lh];
    v8h a1 = *(const v8h*)&As[mw + lm][16 * lh + 8];
    v8h b0 = *(const v8h*)&Bs[nw + lm][16 * lh];
    v8h b1 = *(const v8h*)&Bs[nw + lm][16 * lh + 8];
    v16h af = __builtin_shufflevector(a0, a1, 0,1,2,3,4,5,6,7,8,9,10,11,12,13,14,15);
    v16h bf = __builtin_shufflevector(b0, b1, 0,1,2,3,4,5,6,7,8,9,10,11,12,13,14,15);
    acc = __builtin_amdgcn_wmma_f32_16x16x32_f16(
        false, af, false, bf, (short)0, acc, false, false);
    __syncthreads();
  }

  // ---- epilogue: C VGPR r -> M = r + 8*lh, N = lm --------------------------
  #pragma unroll
  for (int r = 0; r < 8; ++r) {
    int gm = mBase + mw + r + (lh ? 8 : 0);
    int gn = nBase + nw + lm;
    if (gm < M && gn < N) {
      float v = acc[r];
      if (bias) v += bias[gn];
      if (ACT == 1) v = fmaxf(v, 0.f);
      if (OUTH) ((_Float16*)Cv)[cOff + (long long)gm * ldc + gn] = (_Float16)v;
      else      ((float*)Cv)[cOff + (long long)gm * ldc + gn] = v;
    }
  }
}

// ---------------------------------------------------------------------------
// Blockwise random sampling: coords [B,N,2], pf (f32) + pfh (f16) [B,N,C]
// ---------------------------------------------------------------------------
__global__ __launch_bounds__(256)
void sample_kernel(const float* __restrict__ img, const float* __restrict__ brand,
                   float* __restrict__ coords, float* __restrict__ pf,
                   _Float16* __restrict__ pfh)
{
  int bp = blockIdx.x;
  int b = bp / NPTS, p = bp % NPTS;
  int ph = p >> 6;
  int rem = p & 63;
  int pw = rem >> 1, kk = rem & 1;
  const float bs = 2.f / 32.f;

  long long rbase = ((((long long)b * 32 + ph) * 32 + pw) * 2 + kk) * 2;
  float gx = brand[rbase + 0] * bs + (-1.f + ph * bs);
  float gy = brand[rbase + 1] * bs + (-1.f + pw * bs);

  if (threadIdx.x == 0) {
    coords[((long long)b * NPTS + p) * 2 + 0] = gx;
    coords[((long long)b * NPTS + p) * 2 + 1] = gy;
  }
  Bilin bl = bilin_setup(gx, gy, HH, WW2);
  long long ibase = (long long)b * CC * TT;
  long long obase = ((long long)b * NPTS + p) * CC;
  for (int c = threadIdx.x; c < CC; c += 256) {
    long long cb = ibase + (long long)c * TT;
    float v = bl.w[0] * img[cb + bl.off[0]] + bl.w[1] * img[cb + bl.off[1]]
            + bl.w[2] * img[cb + bl.off[2]] + bl.w[3] * img[cb + bl.off[3]];
    pf[obase + c] = v;
    pfh[obase + c] = (_Float16)v;
  }
}

// ---------------------------------------------------------------------------
// Score head: score[r] = sigmoid(dot(hidden[r,:], w2) + b2)
// ---------------------------------------------------------------------------
__global__ __launch_bounds__(256)
void score_head_kernel(const float* __restrict__ hid, const float* __restrict__ w2,
                       const float* __restrict__ b2, float* __restrict__ score)
{
  __shared__ float red[256];
  long long row = blockIdx.x;
  int tid = threadIdx.x;
  float s = 0.f;
  #pragma unroll
  for (int i = 0; i < 3; ++i) {
    int c = tid + i * 256;
    s += hid[row * CC + c] * w2[c];
  }
  s = blk_sum(s, red);
  if (tid == 0) score[row] = 1.f / (1.f + expf(-(s + b2[0])));
}

// ---------------------------------------------------------------------------
// Top-5 selection + gather.  grid: B blocks.
// ---------------------------------------------------------------------------
__global__ __launch_bounds__(256)
void topk_gather_kernel(const float* __restrict__ score, const float* __restrict__ coords,
                        const float* __restrict__ pf,
                        float* __restrict__ px, float* __restrict__ pfk)
{
  __shared__ float s[NPTS];
  __shared__ float rv[256];
  __shared__ int   ri[256];
  __shared__ int   topi[TOPK5];
  int b = blockIdx.x, tid = threadIdx.x;

  for (int p = tid; p < NPTS; p += 256) s[p] = score[(long long)b * NPTS + p];
  __syncthreads();

  for (int it = 0; it < TOPK5; ++it) {
    float mv = -3.4e38f; int mi = 0;
    for (int p = tid; p < NPTS; p += 256) {
      float v = s[p];
      if (v > mv || (v == mv && p < mi)) { mv = v; mi = p; }
    }
    rv[tid] = mv; ri[tid] = mi; __syncthreads();
    #pragma unroll
    for (int st = 128; st > 0; st >>= 1) {
      if (tid < st) {
        if (rv[tid + st] > rv[tid] ||
            (rv[tid + st] == rv[tid] && ri[tid + st] < ri[tid])) {
          rv[tid] = rv[tid + st]; ri[tid] = ri[tid + st];
        }
      }
      __syncthreads();
    }
    if (tid == 0) { topi[it] = ri[0]; s[ri[0]] = -3.4e38f; }
    __syncthreads();
  }

  for (int it = 0; it < TOPK5; ++it) {
    int idx = topi[it];
    if (tid < 2) px[((long long)b * TOPK5 + it) * 2 + tid] =
        coords[((long long)b * NPTS + idx) * 2 + tid];
    for (int c = tid; c < CC; c += 256)
      pfk[((long long)b * TOPK5 + it) * CC + c] =
          pf[((long long)b * NPTS + idx) * CC + c];
  }
}

// ---------------------------------------------------------------------------
// soft_align + modality query, writing f16 rows of mem[B, 10, C]
// ---------------------------------------------------------------------------
__global__ __launch_bounds__(256)
void soft_align_kernel(const float* __restrict__ mainx, const float* __restrict__ pe,
                       const float* __restrict__ px, const float* __restrict__ pfk,
                       const float* __restrict__ arand, const float* __restrict__ mqrow,
                       _Float16* __restrict__ memout, int rowOff)
{
  __shared__ float red[256];
  __shared__ float sw[ARND][4];
  __shared__ int   soff[ARND][4];
  __shared__ float sscore[ARND];
  __shared__ float swt[ARND];

  int bi = blockIdx.x;
  int b = bi / TOPK5, i = bi % TOPK5;
  int tid = threadIdx.x;
  const float* pfr = pfk + ((long long)b * TOPK5 + i) * CC;

  float acc = 0.f;
  #pragma unroll
  for (int q = 0; q < 3; ++q) { float v = pfr[tid + q * 256]; acc += v * v; }
  float pfsq = blk_sum(acc, red);
  float nrm_pf = fmaxf(sqrtf(pfsq), 1e-8f);

  for (int a = 0; a < ARND; ++a) {
    if (tid == 0) {
      float px0 = px[((long long)b * TOPK5 + i) * 2 + 0];
      float px1 = px[((long long)b * TOPK5 + i) * 2 + 1];
      long long ab = (((long long)b * ARND + a) * TOPK5 + i) * 2;
      float gx = px0 + (arand[ab + 0] * 2.f - 0.5f) * 2.f * 0.1f;
      float gy = px1 + (arand[ab + 1] * 2.f - 0.5f) * 2.f * 0.1f;
      gx = fminf(fmaxf(gx, -1.f), 1.f);
      gy = fminf(fmaxf(gy, -1.f), 1.f);
      Bilin bl = bilin_setup(gx, gy, HH, WW2);
      #pragma unroll
      for (int q = 0; q < 4; ++q) { sw[a][q] = bl.w[q]; soff[a][q] = bl.off[q]; }
    }
    __syncthreads();
    float d = 0.f, s2 = 0.f;
    long long ibase = (long long)b * CC * TT;
    #pragma unroll
    for (int q = 0; q < 3; ++q) {
      int c = tid + q * 256;
      long long cb = ibase + (long long)c * TT;
      float af = sw[a][0] * mainx[cb + soff[a][0]] + sw[a][1] * mainx[cb + soff[a][1]]
               + sw[a][2] * mainx[cb + soff[a][2]] + sw[a][3] * mainx[cb + soff[a][3]];
      d += pfr[c] * af;
      s2 += af * af;
    }
    d  = blk_sum(d, red);
    s2 = blk_sum(s2, red);
    if (tid == 0) sscore[a] = d / (nrm_pf * fmaxf(sqrtf(s2), 1e-8f));
    __syncthreads();
  }

  if (tid == 0) {
    float mx = sscore[0];
    for (int a = 1; a < ARND; ++a) mx = fmaxf(mx, sscore[a]);
    float sum = 0.f;
    for (int a = 0; a < ARND; ++a) { swt[a] = expf(sscore[a] - mx); sum += swt[a]; }
    for (int a = 0; a < ARND; ++a) swt[a] /= sum;
  }
  __syncthreads();

  long long orow = ((long long)b * (NMOD * TOPK5) + rowOff + i) * CC;
  #pragma unroll
  for (int q = 0; q < 3; ++q) {
    int c = tid + q * 256;
    float o = pfr[c];
    for (int a = 0; a < ARND; ++a) {
      float emb = sw[a][0] * pe[(long long)soff[a][0] * CC + c]
                + sw[a][1] * pe[(long long)soff[a][1] * CC + c]
                + sw[a][2] * pe[(long long)soff[a][2] * CC + c]
                + sw[a][3] * pe[(long long)soff[a][3] * CC + c];
      o += swt[a] * emb;
    }
    memout[orow + c] = (_Float16)(o + mqrow[c]);
  }
}

// ---------------------------------------------------------------------------
// tgt[b,t,c] = mainx[b,c,t] + pe[t,c] + mq0[c]    (f32 + f16 copies)
// ---------------------------------------------------------------------------
__global__ __launch_bounds__(256)
void build_tgt_kernel(const float* __restrict__ mainx, const float* __restrict__ pe,
                      const float* __restrict__ mq0, float* __restrict__ tgt,
                      _Float16* __restrict__ tgth)
{
  long long total = (long long)BB * TT * CC;
  for (long long idx = (long long)blockIdx.x * 256 + threadIdx.x; idx < total;
       idx += (long long)gridDim.x * 256) {
    int c = (int)(idx % CC);
    long long bt = idx / CC;
    int t = (int)(bt % TT);
    long long b = bt / TT;
    float v = mainx[(b * CC + c) * TT + t] + pe[(long long)t * CC + c] + mq0[c];
    tgt[idx] = v;
    tgth[idx] = (_Float16)v;
  }
}

// ---------------------------------------------------------------------------
// Row softmax with pre-scale; reads f32 scores, writes f16 probabilities.
// ---------------------------------------------------------------------------
__global__ __launch_bounds__(256)
void softmax_kernel(const float* __restrict__ S, _Float16* __restrict__ P, float scale)
{
  __shared__ float red[256];
  long long row = blockIdx.x;
  const float* r = S + row * TT;
  _Float16* w = P + row * TT;
  int tid = threadIdx.x;
  float x[4];
  float mx = -3.4e38f;
  #pragma unroll
  for (int i = 0; i < 4; ++i) { x[i] = r[tid + i * 256] * scale; mx = fmaxf(mx, x[i]); }
  mx = blk_max(mx, red);
  float sum = 0.f;
  #pragma unroll
  for (int i = 0; i < 4; ++i) { x[i] = expf(x[i] - mx); sum += x[i]; }
  sum = blk_sum(sum, red);
  float inv = 1.f / sum;
  #pragma unroll
  for (int i = 0; i < 4; ++i) w[tid + i * 256] = (_Float16)(x[i] * inv);
}

// ---------------------------------------------------------------------------
// out = LayerNorm(a + b) * g + beta; writes f32 out (in-place OK) + f16 copy
// ---------------------------------------------------------------------------
__global__ __launch_bounds__(256)
void add_ln_kernel(const float* __restrict__ a, const float* __restrict__ b,
                   const float* __restrict__ g, const float* __restrict__ be,
                   float* __restrict__ out, _Float16* __restrict__ outh)
{
  __shared__ float red[256];
  long long row = blockIdx.x;
  int tid = threadIdx.x;
  float x[3];
  #pragma unroll
  for (int i = 0; i < 3; ++i) {
    int c = tid + i * 256;
    float v = a[row * CC + c];
    if (b) v += b[row * CC + c];
    x[i] = v;
  }
  float mean = blk_sum(x[0] + x[1] + x[2], red) * (1.f / CC);
  float q = 0.f;
  #pragma unroll
  for (int i = 0; i < 3; ++i) { float d = x[i] - mean; q += d * d; }
  float var = blk_sum(q, red) * (1.f / CC);
  float inv = rsqrtf(var + 1e-5f);
  #pragma unroll
  for (int i = 0; i < 3; ++i) {
    int c = tid + i * 256;
    float v = (x[i] - mean) * inv * g[c] + be[c];
    out[row * CC + c] = v;
    outh[row * CC + c] = (_Float16)v;
  }
}

// ---------------------------------------------------------------------------
// Cross-attention core: 10 keys, 8 heads, hd=96.  f16 in / f16 out.
// ---------------------------------------------------------------------------
__global__ __launch_bounds__(256)
void cross_attn_kernel(const _Float16* __restrict__ q, const _Float16* __restrict__ k,
                       const _Float16* __restrict__ v, _Float16* __restrict__ o)
{
  __shared__ float sc[NHEAD][10];
  long long bt = blockIdx.x;
  int b = (int)(bt >> 10);
  int tid = threadIdx.x;
  const _Float16* qr = q + bt * CC;
  const float scale = 0.10206207261596575f;  // 1/sqrt(96)

  if (tid < NHEAD * 10) {
    int h = tid / 10, j = tid % 10;
    const _Float16* kr = k + ((long long)b * 10 + j) * CC + h * HD;
    const _Float16* qh = qr + h * HD;
    float s = 0.f;
    #pragma unroll 8
    for (int d = 0; d < HD; ++d) s += (float)qh[d] * (float)kr[d];
    sc[h][j] = s * scale;
  }
  __syncthreads();
  if (tid < NHEAD) {
    float mx = sc[tid][0];
    for (int j = 1; j < 10; ++j) mx = fmaxf(mx, sc[tid][j]);
    float sum = 0.f;
    for (int j = 0; j < 10; ++j) { sc[tid][j] = expf(sc[tid][j] - mx); sum += sc[tid][j]; }
    float inv = 1.f / sum;
    for (int j = 0; j < 10; ++j) sc[tid][j] *= inv;
  }
  __syncthreads();
  #pragma unroll
  for (int qq = 0; qq < 3; ++qq) {
    int c = tid + qq * 256;
    int h = c / HD;
    float acc = 0.f;
    for (int j = 0; j < 10; ++j)
      acc += sc[h][j] * (float)v[((long long)b * 10 + j) * CC + c];
    o[bt * CC + c] = (_Float16)acc;
  }
}

// ---------------------------------------------------------------------------
// out[b,c,h,w] = tgt[b, h*W+w, c]
// ---------------------------------------------------------------------------
__global__ __launch_bounds__(256)
void final_transpose_kernel(const float* __restrict__ tgt, float* __restrict__ out)
{
  long long total = (long long)BB * TT * CC;
  for (long long idx = (long long)blockIdx.x * 256 + threadIdx.x; idx < total;
       idx += (long long)gridDim.x * 256) {
    int c = (int)(idx % CC);
    long long bt = idx / CC;
    int t = (int)(bt % TT);
    long long b = bt / TT;
    out[(b * CC + c) * TT + t] = tgt[idx];
  }
}

// ---------------------------------------------------------------------------
// Host-side orchestration
// ---------------------------------------------------------------------------
static inline dim3 gemm_grid(int M, int N, int batch) {
  return dim3((N + 63) / 64, (M + 31) / 32, batch);
}

static inline void cvt(const float* in, _Float16* out, long long n, hipStream_t s) {
  long long blocks = (n / 4 + 255) / 256;
  if (blocks > 4096) blocks = 4096;
  cvt_h_kernel<<<(int)blocks, 256, 0, s>>>(in, out, n);
}

extern "C" void kernel_launch(void* const* d_in, const int* in_sizes, int n_in,
                              void* d_out, int out_size, void* d_ws, size_t ws_size,
                              hipStream_t stream) {
  (void)in_sizes; (void)n_in; (void)out_size; (void)ws_size;

  const float* mainx    = (const float*)d_in[0];
  const float* others[2]= {(const float*)d_in[1], (const float*)d_in[2]};
  const float* pe       = (const float*)d_in[3];
  const float* mq       = (const float*)d_in[4];
  const float* s_w1     = (const float*)d_in[5];
  const float* s_b1     = (const float*)d_in[6];
  const float* s_w2     = (const float*)d_in[7];
  const float* s_b2     = (const float*)d_in[8];
  const float* sa_in_w  = (const float*)d_in[9];
  const float* sa_in_b  = (const float*)d_in[10];
  const float* sa_out_w = (const float*)d_in[11];
  const float* sa_out_b = (const float*)d_in[12];
  const float* ca_in_w  = (const float*)d_in[13];
  const float* ca_in_b  = (const float*)d_in[14];
  const float* ca_out_w = (const float*)d_in[15];
  const float* ca_out_b = (const float*)d_in[16];
  const float* ff1_w    = (const float*)d_in[17];
  const float* ff1_b    = (const float*)d_in[18];
  const float* ff2_w    = (const float*)d_in[19];
  const float* ff2_b    = (const float*)d_in[20];
  const float* ln_w     = (const float*)d_in[21];
  const float* ln_b     = (const float*)d_in[22];
  const float* block_rand  = (const float*)d_in[23];
  const float* around_rand = (const float*)d_in[24];
  float* out = (float*)d_out;

  // ---- workspace carve-out (16B-aligned slabs) ----------------------------
  char* wsb = (char*)d_ws;
  size_t off = 0;
  auto allocf = [&](size_t n) {
    float* p = (float*)(wsb + off); off = (off + n * 4 + 15) & ~(size_t)15; return p; };
  auto alloch = [&](size_t n) {
    _Float16* p = (_Float16*)(wsb + off); off = (off + n * 2 + 15) & ~(size_t)15; return p; };

  float*    pf     = allocf((size_t)BB * NPTS * CC);
  _Float16* pfh    = alloch((size_t)BB * NPTS * CC);
  float*    hid    = allocf((size_t)BB * NPTS * CC);
  float*    score  = allocf((size_t)BB * NPTS);
  float*    coords = allocf((size_t)BB * NPTS * 2);
  float*    px     = allocf((size_t)BB * TOPK5 * 2);
  float*    pfk    = allocf((size_t)BB * TOPK5 * CC);
  _Float16* memBh  = alloch((size_t)BB * NMOD * TOPK5 * CC);
  float*    tgt    = allocf((size_t)BB * TT * CC);
  _Float16* tgth   = alloch((size_t)BB * TT * CC);
  _Float16* qkvh   = alloch((size_t)BB * TT * 3 * CC);
  float*    Sbuf   = allocf((size_t)BB * NHEAD * TT * TT);
  _Float16* Ph     = alloch((size_t)BB * NHEAD * TT * TT);
  _Float16* attnoh = alloch((size_t)BB * TT * CC);
  float*    proj   = allocf((size_t)BB * TT * CC);
  _Float16* qbufh  = alloch((size_t)BB * TT * CC);
  _Float16* kbufh  = alloch((size_t)BB * NMOD * TOPK5 * CC);
  _Float16* vbufh  = alloch((size_t)BB * NMOD * TOPK5 * CC);
  _Float16* ffhh   = alloch((size_t)BB * TT * FFD);
  float*    ffo    = allocf((size_t)BB * TT * CC);
  // f16 weight copies
  _Float16* s_w1h    = alloch((size_t)CC * CC);
  _Float16* sa_in_wh = alloch((size_t)LNUM * CC * 3 * CC);
  _Float16* sa_out_wh= alloch((size_t)LNUM * CC * CC);
  _Float16* ca_in_wh = alloch((size_t)LNUM * CC * 3 * CC);
  _Float16* ca_out_wh= alloch((size_t)LNUM * CC * CC);
  _Float16* ff1_wh   = alloch((size_t)LNUM * CC * FFD);
  _Float16* ff2_wh   = alloch((size_t)LNUM * FFD * CC);

  const int MT  = BB * TT;             // 8192
  const int MS  = BB * NPTS;           // 16384
  const int MKV = BB * NMOD * TOPK5;   // 80

  // ---- weight conversions -------------------------------------------------
  cvt(s_w1,     s_w1h,     (long long)CC * CC, stream);
  cvt(sa_in_w,  sa_in_wh,  (long long)LNUM * CC * 3 * CC, stream);
  cvt(sa_out_w, sa_out_wh, (long long)LNUM * CC * CC, stream);
  cvt(ca_in_w,  ca_in_wh,  (long long)LNUM * CC * 3 * CC, stream);
  cvt(ca_out_w, ca_out_wh, (long long)LNUM * CC * CC, stream);
  cvt(ff1_w,    ff1_wh,    (long long)LNUM * CC * FFD, stream);
  cvt(ff2_w,    ff2_wh,    (long long)LNUM * FFD * CC, stream);

  // ---- per-modality sampling / scoring / alignment ------------------------
  for (int mi = 0; mi < NMOD; ++mi) {
    const float* brand = block_rand + (size_t)mi * BB * 32 * 32 * 2 * 2;
    const float* arand = around_rand + (size_t)mi * BB * ARND * TOPK5 * 2;

    sample_kernel<<<BB * NPTS, 256, 0, stream>>>(others[mi], brand, coords, pf, pfh);

    gemm_wmma<0, 1, 0><<<gemm_grid(MS, CC, 1), 256, 0, stream>>>(
        pfh, CC, 0, 0, s_w1h, CC, 0, 0, hid, CC, 0, 0, s_b1, MS, CC, CC, 1);

    score_head_kernel<<<MS, 256, 0, stream>>>(hid, s_w2, s_b2, score);
    topk_gather_kernel<<<BB, 256, 0, stream>>>(score, coords, pf, px, pfk);
    soft_align_kernel<<<BB * TOPK5, 256, 0, stream>>>(
        mainx, pe, px, pfk, arand, mq + (size_t)(mi + 1) * CC, memBh, mi * TOPK5);
  }

  // ---- tgt = mainx^T + mq[0] + pe ----------------------------------------
  build_tgt_kernel<<<8192, 256, 0, stream>>>(mainx, pe, mq, tgt, tgth);

  // ---- decoder layers -----------------------------------------------------
  const float rs96 = 0.10206207261596575f;
  for (int l = 0; l < LNUM; ++l) {
    const _Float16* wsa_i = sa_in_wh  + (size_t)l * CC * 3 * CC;
    const float*    bsa_i = sa_in_b   + (size_t)l * 3 * CC;
    const _Float16* wsa_o = sa_out_wh + (size_t)l * CC * CC;
    const float*    bsa_o = sa_out_b  + (size_t)l * CC;
    const _Float16* wca_i = ca_in_wh  + (size_t)l * CC * 3 * CC;
    const float*    bca_i = ca_in_b   + (size_t)l * 3 * CC;
    const _Float16* wca_o = ca_out_wh + (size_t)l * CC * CC;
    const float*    bca_o = ca_out_b  + (size_t)l * CC;
    const _Float16* w1    = ff1_wh + (size_t)l * CC * FFD;
    const float*    b1    = ff1_b  + (size_t)l * FFD;
    const _Float16* w2    = ff2_wh + (size_t)l * FFD * CC;
    const float*    b2    = ff2_b  + (size_t)l * CC;
    const float*    lg    = ln_w + (size_t)l * 3 * CC;
    const float*    lb    = ln_b + (size_t)l * 3 * CC;

    // ---- self attention ----
    gemm_wmma<0, 0, 1><<<gemm_grid(MT, 3 * CC, 1), 256, 0, stream>>>(
        tgth, CC, 0, 0, wsa_i, 3 * CC, 0, 0, qkvh, 3 * CC, 0, 0, bsa_i,
        MT, 3 * CC, CC, 1);
    // S[b,h] = Q @ K^T
    gemm_wmma<1, 0, 0><<<gemm_grid(TT, TT, BB * NHEAD), 256, 0, stream>>>(
        qkvh, 3 * CC, (long long)TT * 3 * CC, HD,
        qkvh + CC, 3 * CC, (long long)TT * 3 * CC, HD,
        Sbuf, TT, (long long)NHEAD * TT * TT, (long long)TT * TT,
        nullptr, TT, TT, HD, NHEAD);
    softmax_kernel<<<BB * NHEAD * TT, 256, 0, stream>>>(Sbuf, Ph, rs96);
    // O[b,h] = P @ V
    gemm_wmma<0, 0, 1><<<gemm_grid(TT, HD, BB * NHEAD), 256, 0, stream>>>(
        Ph, TT, (long long)NHEAD * TT * TT, (long long)TT * TT,
        qkvh + 2 * CC, 3 * CC, (long long)TT * 3 * CC, HD,
        attnoh, CC, (long long)TT * CC, HD,
        nullptr, TT, HD, TT, NHEAD);
    gemm_wmma<0, 0, 0><<<gemm_grid(MT, CC, 1), 256, 0, stream>>>(
        attnoh, CC, 0, 0, wsa_o, CC, 0, 0, proj, CC, 0, 0, bsa_o, MT, CC, CC, 1);
    add_ln_kernel<<<MT, 256, 0, stream>>>(tgt, proj, lg, lb, tgt, tgth);

    // ---- cross attention ----
    gemm_wmma<0, 0, 1><<<gemm_grid(MT, CC, 1), 256, 0, stream>>>(
        tgth, CC, 0, 0, wca_i, 3 * CC, 0, 0, qbufh, CC, 0, 0, bca_i, MT, CC, CC, 1);
    gemm_wmma<0, 0, 1><<<gemm_grid(MKV, CC, 1), 256, 0, stream>>>(
        memBh, CC, 0, 0, wca_i + CC, 3 * CC, 0, 0, kbufh, CC, 0, 0, bca_i + CC,
        MKV, CC, CC, 1);
    gemm_wmma<0, 0, 1><<<gemm_grid(MKV, CC, 1), 256, 0, stream>>>(
        memBh, CC, 0, 0, wca_i + 2 * CC, 3 * CC, 0, 0, vbufh, CC, 0, 0, bca_i + 2 * CC,
        MKV, CC, CC, 1);
    cross_attn_kernel<<<MT, 256, 0, stream>>>(qbufh, kbufh, vbufh, attnoh);
    gemm_wmma<0, 0, 0><<<gemm_grid(MT, CC, 1), 256, 0, stream>>>(
        attnoh, CC, 0, 0, wca_o, CC, 0, 0, proj, CC, 0, 0, bca_o, MT, CC, CC, 1);
    add_ln_kernel<<<MT, 256, 0, stream>>>(tgt, proj, lg + CC, lb + CC, tgt, tgth);

    // ---- feed-forward ----
    gemm_wmma<0, 1, 1><<<gemm_grid(MT, FFD, 1), 256, 0, stream>>>(
        tgth, CC, 0, 0, w1, FFD, 0, 0, ffhh, FFD, 0, 0, b1, MT, FFD, CC, 1);
    gemm_wmma<0, 0, 0><<<gemm_grid(MT, CC, 1), 256, 0, stream>>>(
        ffhh, FFD, 0, 0, w2, CC, 0, 0, ffo, CC, 0, 0, b2, MT, CC, FFD, 1);
    add_ln_kernel<<<MT, 256, 0, stream>>>(tgt, ffo, lg + 2 * CC, lb + 2 * CC, tgt, tgth);
  }

  final_transpose_kernel<<<8192, 256, 0, stream>>>(tgt, out);
}
